// DMSTransformerBlock_10222022164495
// MI455X (gfx1250) — matrix operand order repaired and checked by reading, not compiled, not run on hardware
//
#include <hip/hip_runtime.h>
#include <hip/hip_bf16.h>
#include <math.h>

// ---------------- problem constants ----------------
#define Bn    8
#define Cch   256
#define Hh    14
#define Ww    14
#define NHW   196      // H*W
#define Ssz   49       // RH*RW
#define RHh   7
#define RWw   7
#define HEADS 8
#define HCc   32
#define NS    (NHW*Ssz)   // 9604

typedef _Float16 half16 __attribute__((ext_vector_type(16)));
typedef _Float16 half8  __attribute__((ext_vector_type(8)));
typedef float    float8 __attribute__((ext_vector_type(8)));

__device__ __forceinline__ float gelu_f(float x) {
    return 0.5f * x * (1.0f + erff(x * 0.70710678118654752f));
}
__device__ __forceinline__ int iclamp(int v, int lo, int hi) {
    return v < lo ? lo : (v > hi ? hi : v);
}

// ---------------- prep: transpose x to channel-last, f32 + f16 ----------------
__global__ __launch_bounds__(256) void prep_x_kernel(const float* __restrict__ x,
                                                     float* __restrict__ xt,
                                                     _Float16* __restrict__ xt_h) {
    int idx = blockIdx.x * 256 + threadIdx.x;
    if (idx >= Bn * Cch * NHW) return;
    int b = idx / (Cch * NHW);
    int r = idx % (Cch * NHW);
    int c = r / NHW;
    int p = r % NHW;
    float v = x[idx];
    size_t o = ((size_t)b * NHW + p) * Cch + c;
    xt[o] = v;
    xt_h[o] = (_Float16)v;
}

// ---------------- prep: weights f32 -> f16 ----------------
__global__ __launch_bounds__(256) void prep_w_kernel(
    const float* s0, const float* s1, const float* s2, const float* s3, const float* s4,
    _Float16* d0, _Float16* d1, _Float16* d2, _Float16* d3, _Float16* d4) {
    int idx = blockIdx.x * 256 + threadIdx.x;
    int w = idx / (Cch * Cch);
    int j = idx % (Cch * Cch);
    const float* s = nullptr; _Float16* d = nullptr;
    switch (w) {
        case 0: s = s0; d = d0; break;
        case 1: s = s1; d = d1; break;
        case 2: s = s2; d = d2; break;
        case 3: s = s3; d = d3; break;
        case 4: s = s4; d = d4; break;
        default: return;
    }
    d[j] = (_Float16)s[j];
}

// ---------------- WMMA GEMM: D[m,col] = sum_k A[m,k]*B[col,k] + bias[m] ----------------
// A: f16 [256x256] row-major (M x K).  B: f16 channel-last, per batch [Ncol x 256].
// One wave computes a 16x64 output tile (4 x 16x16 WMMA accumulators) over K=256
// in 8 steps of 32: A fetched once per K-step, reused by 4 WMMAs.
// Out-of-range columns load a clamped valid column (no EXEC branching); their
// results are simply never stored.
// EPI: 0 = bias, write f32 + f16 channel-last
//      1 = bias + GELU, write f32 channel-last
//      2 = bias, write f16 channel-last
//      3 = bias, write f32 CHANNEL-FIRST (final output (B,C,H,W))
template <int EPI>
__global__ __launch_bounds__(256) void wmma_gemm_kernel(
    const _Float16* __restrict__ A, const _Float16* __restrict__ Bm,
    const float* __restrict__ bias, float* __restrict__ outF,
    _Float16* __restrict__ outH, int Ncol, int ctiles64, int totalWaves) {
    int wid = blockIdx.x * 8 + (threadIdx.x >> 5);
    if (wid >= totalWaves) return;               // wave-uniform
    int lane = threadIdx.x & 31;
    int b   = wid / (16 * ctiles64);
    int rem = wid % (16 * ctiles64);
    int mt  = rem / ctiles64;
    int ct  = rem % ctiles64;

    int hk = lane >> 4;                          // half-wave selector (0/1)
    // A: lane holds row m; halves 0..15 = K kb + hk*16 + 0..15
    int m = mt * 16 + (lane & 15);
    const _Float16* abase = A + (size_t)m * Cch + hk * 16;

    // B: lane j-th subtile column; halves 0..15 = K kb + hk*16 + 0..15 (32B aligned)
    int col0 = ct * 64 + (lane & 15);
    int c0 = col0 < Ncol ? col0 : (Ncol - 1);
    int c1 = (col0 + 16) < Ncol ? (col0 + 16) : (Ncol - 1);
    int c2 = (col0 + 32) < Ncol ? (col0 + 32) : (Ncol - 1);
    int c3 = (col0 + 48) < Ncol ? (col0 + 48) : (Ncol - 1);
    size_t bb = (size_t)b * Ncol;
    const _Float16* bbase0 = Bm + (bb + c0) * Cch + hk * 16;
    const _Float16* bbase1 = Bm + (bb + c1) * Cch + hk * 16;
    const _Float16* bbase2 = Bm + (bb + c2) * Cch + hk * 16;
    const _Float16* bbase3 = Bm + (bb + c3) * Cch + hk * 16;

    float8 acc0 = {}, acc1 = {}, acc2 = {}, acc3 = {};
#pragma unroll
    for (int kb = 0; kb < Cch; kb += 32) {
        half8 alo = *(const half8*)(abase + kb);
        half8 ahi = *(const half8*)(abase + kb + 8);
        half16 av;
#pragma unroll
        for (int i = 0; i < 8; ++i) { av[i] = alo[i]; av[8 + i] = ahi[i]; }
        half16 bv0 = *(const half16*)(bbase0 + kb);
        half16 bv1 = *(const half16*)(bbase1 + kb);
        half16 bv2 = *(const half16*)(bbase2 + kb);
        half16 bv3 = *(const half16*)(bbase3 + kb);
        acc0 = __builtin_amdgcn_wmma_f32_16x16x32_f16(false, av, false, bv0, (short)0, acc0, false, false);
        acc1 = __builtin_amdgcn_wmma_f32_16x16x32_f16(false, av, false, bv1, (short)0, acc1, false, false);
        acc2 = __builtin_amdgcn_wmma_f32_16x16x32_f16(false, av, false, bv2, (short)0, acc2, false, false);
        acc3 = __builtin_amdgcn_wmma_f32_16x16x32_f16(false, av, false, bv3, (short)0, acc3, false, false);
    }

    int rowbase = mt * 16 + hk * 8;              // D: lane<16 rows 0..7, lane>=16 rows 8..15
    auto store_tile = [&](float8 accv, int colj) {
        if (colj >= Ncol) return;
        size_t obase = ((size_t)b * Ncol + colj) * Cch + rowbase;
#pragma unroll
        for (int i = 0; i < 8; ++i) {
            float v = accv[i] + bias[rowbase + i];
            if (EPI == 1) v = gelu_f(v);
            if (EPI == 0) { outF[obase + i] = v; outH[obase + i] = (_Float16)v; }
            if (EPI == 1) { outF[obase + i] = v; }
            if (EPI == 2) { outH[obase + i] = (_Float16)v; }
            if (EPI == 3) { outF[((size_t)b * Cch + rowbase + i) * Ncol + colj] = v; }
        }
    };
    store_tile(acc0, col0);
    store_tile(acc1, col0 + 16);
    store_tile(acc2, col0 + 32);
    store_tile(acc3, col0 + 48);
}

// ---------------- depthwise 3x3 stride2 + LayerNorm(C) + GELU ----------------
// in: t1 channel-last (B,NHW,C) f32 ; out: t2 channel-last (B,S,C) f32
__global__ __launch_bounds__(256) void dwln_kernel(
    const float* __restrict__ t1, const float* __restrict__ dw_w,
    const float* __restrict__ dw_b, const float* __restrict__ ln_g,
    const float* __restrict__ ln_b, float* __restrict__ t2) {
    int brs = blockIdx.x;                 // b*49 + rs
    int b = brs / Ssz, rs = brs % Ssz;
    int c = threadIdx.x;
    int ry = rs / RWw, rx = rs % RWw;
    float acc = dw_b[c];
#pragma unroll
    for (int ky = 0; ky < 3; ++ky) {
        int y = ry * 2 - 1 + ky;
        if (y < 0 || y >= Hh) continue;
#pragma unroll
        for (int kx = 0; kx < 3; ++kx) {
            int xx = rx * 2 - 1 + kx;
            if (xx < 0 || xx >= Ww) continue;
            acc += t1[((size_t)b * NHW + y * Ww + xx) * Cch + c] * dw_w[c * 9 + ky * 3 + kx];
        }
    }
    __shared__ float red[256];
    red[c] = acc; __syncthreads();
    for (int s = 128; s > 0; s >>= 1) { if (c < s) red[c] += red[c + s]; __syncthreads(); }
    float mu = red[0] * (1.0f / 256.0f);
    __syncthreads();
    float d = acc - mu;
    red[c] = d * d; __syncthreads();
    for (int s = 128; s > 0; s >>= 1) { if (c < s) red[c] += red[c + s]; __syncthreads(); }
    float var = red[0] * (1.0f / 256.0f);
    float t = d * rsqrtf(var + 1e-5f) * ln_g[c] + ln_b[c];
    t2[(size_t)brs * Cch + c] = gelu_f(t);
}

// ---------------- off2 (tiny GEMM) + tanh scale -> offsets (B,N,S,2) ----------------
__global__ __launch_bounds__(64) void off2_kernel(const float* __restrict__ off2_w,
                                                  const float* __restrict__ t2,
                                                  float* __restrict__ off) {
    int bo = blockIdx.x;                  // b*392 + o
    int b = bo / (2 * NHW), o = bo % (2 * NHW);
    int s = threadIdx.x;
    if (s >= Ssz) return;
    const float* wrow = off2_w + (size_t)o * Cch;
    const float* trow = t2 + ((size_t)b * Ssz + s) * Cch;
    float acc = 0.0f;
    for (int c = 0; c < Cch; ++c) acc += wrow[c] * trow[c];
    int d = o / NHW, n = o % NHW;
    float scale = (d == 0) ? (2.0f / (float)Hh) : (2.0f / (float)Ww);   // FACTOR/range
    off[(((size_t)b * NHW + n) * Ssz + s) * 2 + d] = tanhf(acc) * scale;
}

// ---------------- bilinear gather: sampled (B,N,S,C) f16, channel-last ----------------
__global__ __launch_bounds__(256) void sample_kernel(const float* __restrict__ xt,
                                                     const float* __restrict__ off,
                                                     _Float16* __restrict__ samp) {
    int bns = blockIdx.x;                 // ((b*196)+n)*49 + s
    int b = bns / (NHW * Ssz);
    int r = bns % (NHW * Ssz);
    int s = r % Ssz;
    int c = threadIdx.x;
    float offy = off[(size_t)bns * 2 + 0];
    float offx = off[(size_t)bns * 2 + 1];
    int sy = s / RWw, sx = s % RWw;
    float gyref = (float)(sy * 2) * (2.0f / 13.0f) - 1.0f;
    float gxref = (float)(sx * 2) * (2.0f / 13.0f) - 1.0f;
    float gy = (offy + gyref + 1.0f) * 6.5f;
    float gx = (offx + gxref + 1.0f) * 6.5f;
    float y0f = floorf(gy), x0f = floorf(gx);
    float wy = gy - y0f, wx = gx - x0f;
    int y0 = (int)y0f, x0 = (int)x0f;
    const float* xb = xt + (size_t)b * NHW * Cch;
    float acc = 0.0f;
    auto corner = [&](int yi, int xi, float wt) {
        bool valid = (yi >= 0) && (yi < Hh) && (xi >= 0) && (xi < Ww);
        int yc = iclamp(yi, 0, Hh - 1), xc = iclamp(xi, 0, Ww - 1);
        float v = xb[((size_t)(yc * Ww + xc)) * Cch + c];
        acc += v * (valid ? wt : 0.0f);
    };
    corner(y0,     x0,     (1.0f - wy) * (1.0f - wx));
    corner(y0,     x0 + 1, (1.0f - wy) * wx);
    corner(y0 + 1, x0,     wy * (1.0f - wx));
    corner(y0 + 1, x0 + 1, wy * wx);
    samp[(size_t)bns * Cch + c] = (_Float16)acc;
}

// ---------------- fused attention: qk + rel-pos bilinear bias + softmax + AV ----------------
__global__ __launch_bounds__(256) void attn_kernel(
    const float* __restrict__ q_cl, const _Float16* __restrict__ kf,
    const _Float16* __restrict__ vf, const float* __restrict__ off,
    const float* __restrict__ posembed, _Float16* __restrict__ outH) {
    int bn = blockIdx.x;                  // b*196 + n
    int n = bn % NHW;
    int t = threadIdx.x;
    __shared__ float qS[Cch];
    __shared__ float attnS[HEADS * Ssz];
    qS[t] = q_cl[(size_t)bn * Cch + t];
    __syncthreads();
    int ny = n / Ww, nx = n % Ww;
    for (int idx = t; idx < HEADS * Ssz; idx += 256) {
        int h = idx / Ssz, s = idx % Ssz;
        // vectorized 32-element dot (two 32B loads, 64B-aligned)
        const half16* kp16 = (const half16*)(kf + ((size_t)bn * Ssz + s) * Cch + h * HCc);
        half16 k0 = kp16[0];
        half16 k1 = kp16[1];
        float acc = 0.0f;
#pragma unroll
        for (int c2 = 0; c2 < 16; ++c2) acc += qS[h * HCc + c2] * (float)k0[c2];
#pragma unroll
        for (int c2 = 0; c2 < 16; ++c2) acc += qS[h * HCc + 16 + c2] * (float)k1[c2];
        // dynamic relative-position bias, bilinear from posembed[h] (27x27), zero-pad
        float offy = off[((size_t)bn * Ssz + s) * 2 + 0];
        float offx = off[((size_t)bn * Ssz + s) * 2 + 1];
        int sy = s / RWw, sx = s % RWw;
        float rpy = (float)(2 * sy - ny) * (1.0f / 13.0f);
        float rpx = (float)(2 * sx - nx) * (1.0f / 13.0f);
        float ty = (rpy - offy + 1.0f) * 13.0f;
        float tx = (rpx - offx + 1.0f) * 13.0f;
        float y0f = floorf(ty), x0f = floorf(tx);
        float wy = ty - y0f, wx = tx - x0f;
        int y0 = (int)y0f, x0 = (int)x0f;
        const float* tab = posembed + (size_t)h * 27 * 27;
        float bias = 0.0f;
        auto tcorner = [&](int yi, int xi, float wt) {
            bool valid = (yi >= 0) && (yi < 27) && (xi >= 0) && (xi < 27);
            int yc = iclamp(yi, 0, 26), xc = iclamp(xi, 0, 26);
            bias += tab[yc * 27 + xc] * (valid ? wt : 0.0f);
        };
        tcorner(y0,     x0,     (1.0f - wy) * (1.0f - wx));
        tcorner(y0,     x0 + 1, (1.0f - wy) * wx);
        tcorner(y0 + 1, x0,     wy * (1.0f - wx));
        tcorner(y0 + 1, x0 + 1, wy * wx);
        attnS[idx] = acc + bias;
    }
    __syncthreads();
    if (t < HEADS) {                      // per-head softmax over S=49
        float m = -1e30f;
        for (int s = 0; s < Ssz; ++s) m = fmaxf(m, attnS[t * Ssz + s]);
        float sum = 0.0f;
        for (int s = 0; s < Ssz; ++s) { float e = expf(attnS[t * Ssz + s] - m); attnS[t * Ssz + s] = e; sum += e; }
        float inv = 1.0f / sum;
        for (int s = 0; s < Ssz; ++s) attnS[t * Ssz + s] *= inv;
    }
    __syncthreads();
    int h = t / HCc;
    const _Float16* vp = vf + (size_t)bn * Ssz * Cch + t;
    float o = 0.0f;
    for (int s = 0; s < Ssz; ++s) o += attnS[h * Ssz + s] * (float)vp[s * Cch];
    outH[(size_t)bn * Cch + t] = (_Float16)o;
}

// ---------------- host launcher ----------------
extern "C" void kernel_launch(void* const* d_in, const int* in_sizes, int n_in,
                              void* d_out, int out_size, void* d_ws, size_t ws_size,
                              hipStream_t stream) {
    const float* x        = (const float*)d_in[0];
    const float* q_w      = (const float*)d_in[1];
    const float* q_b      = (const float*)d_in[2];
    const float* k_w      = (const float*)d_in[3];
    const float* k_b      = (const float*)d_in[4];
    const float* v_w      = (const float*)d_in[5];
    const float* v_b      = (const float*)d_in[6];
    const float* off1_w   = (const float*)d_in[7];
    const float* off1_b   = (const float*)d_in[8];
    const float* dw_w     = (const float*)d_in[9];
    const float* dw_b     = (const float*)d_in[10];
    const float* ln_g     = (const float*)d_in[11];
    const float* ln_b     = (const float*)d_in[12];
    const float* off2_w   = (const float*)d_in[13];
    const float* posembed = (const float*)d_in[14];
    const float* proj_w   = (const float*)d_in[15];
    const float* proj_b   = (const float*)d_in[16];
    float* out = (float*)d_out;

    char* p = (char*)d_ws;
    auto alloc = [&](size_t bytes) -> char* {
        char* r = p; p += (bytes + 255) & ~(size_t)255; return r;
    };
    const size_t EperBNC = (size_t)Bn * NHW * Cch;       // 401408
    float*    xt      = (float*)alloc(EperBNC * 4);
    _Float16* xt_h    = (_Float16*)alloc(EperBNC * 2);
    _Float16* q_wh    = (_Float16*)alloc((size_t)Cch * Cch * 2);
    _Float16* off1_wh = (_Float16*)alloc((size_t)Cch * Cch * 2);
    _Float16* k_wh    = (_Float16*)alloc((size_t)Cch * Cch * 2);
    _Float16* v_wh    = (_Float16*)alloc((size_t)Cch * Cch * 2);
    _Float16* proj_wh = (_Float16*)alloc((size_t)Cch * Cch * 2);
    float*    q_cl    = (float*)alloc(EperBNC * 4);
    _Float16* q_h     = (_Float16*)alloc(EperBNC * 2);
    float*    t1      = (float*)alloc(EperBNC * 4);
    float*    t2      = (float*)alloc((size_t)Bn * Ssz * Cch * 4);
    float*    offb    = (float*)alloc((size_t)Bn * NHW * Ssz * 2 * 4);
    _Float16* samp    = (_Float16*)alloc((size_t)Bn * NS * Cch * 2);
    _Float16* kfb     = (_Float16*)alloc((size_t)Bn * NS * Cch * 2);
    _Float16* vfb     = (_Float16*)alloc((size_t)Bn * NS * Cch * 2);
    _Float16* outh    = (_Float16*)alloc(EperBNC * 2);

    // 1) transpose/convert x, convert weights
    prep_x_kernel<<<(int)((EperBNC + 255) / 256), 256, 0, stream>>>(x, xt, xt_h);
    prep_w_kernel<<<(5 * Cch * Cch) / 256, 256, 0, stream>>>(
        q_w, off1_w, k_w, v_w, proj_w, q_wh, off1_wh, k_wh, v_wh, proj_wh);

    // 2) q = q_w@x + q_b  (channel-last f32 + f16)
    {
        int ct = (NHW + 63) / 64, tw = Bn * 16 * ct;
        wmma_gemm_kernel<0><<<(tw + 7) / 8, 256, 0, stream>>>(
            q_wh, xt_h, q_b, q_cl, q_h, NHW, ct, tw);
    }
    // 3) t1 = GELU(off1_w@q + off1_b)
    {
        int ct = (NHW + 63) / 64, tw = Bn * 16 * ct;
        wmma_gemm_kernel<1><<<(tw + 7) / 8, 256, 0, stream>>>(
            off1_wh, q_h, off1_b, t1, nullptr, NHW, ct, tw);
    }
    // 4) depthwise conv + LN + GELU
    dwln_kernel<<<Bn * Ssz, 256, 0, stream>>>(t1, dw_w, dw_b, ln_g, ln_b, t2);
    // 5) offsets
    off2_kernel<<<Bn * 2 * NHW, 64, 0, stream>>>(off2_w, t2, offb);
    // 6) bilinear gather
    sample_kernel<<<Bn * NHW * Ssz, 256, 0, stream>>>(xt, offb, samp);
    // 7) kf / vf projections of sampled features (the big WMMA GEMMs)
    {
        int ct = (NS + 63) / 64, tw = Bn * 16 * ct;
        wmma_gemm_kernel<2><<<(tw + 7) / 8, 256, 0, stream>>>(
            k_wh, samp, k_b, nullptr, kfb, NS, ct, tw);
        wmma_gemm_kernel<2><<<(tw + 7) / 8, 256, 0, stream>>>(
            v_wh, samp, v_b, nullptr, vfb, NS, ct, tw);
    }
    // 8) fused attention
    attn_kernel<<<Bn * NHW, 256, 0, stream>>>(q_cl, kfb, vfb, offb, posembed, outh);
    // 9) final projection, channel-first into d_out
    {
        int ct = (NHW + 63) / 64, tw = Bn * 16 * ct;
        wmma_gemm_kernel<3><<<(tw + 7) / 8, 256, 0, stream>>>(
            proj_wh, outh, proj_b, out, nullptr, NHW, ct, tw);
    }
}
